// ConstrastiveLoss_69999376990701
// MI455X (gfx1250) — matrix operand order repaired and checked by reading, not compiled, not run on hardware
//
#include <hip/hip_runtime.h>
#include <hip/hip_bf16.h>
#include <stdint.h>

#define V_ 4
#define B_ 2048
#define D_ 512
#define N_ (V_*B_)
#define INV_TEMP 2.0f
#define L2E 1.4426950408889634f
#define LN2 0.6931471805599453f

#define ROWS_WG 128
#define BCOLS 32                    /* B stripe width (2 WMMA tiles per wave) */
#define ABYTES (ROWS_WG*D_*2)       /* 128 KB A tile in LDS  */
#define BBYTES (BCOLS*D_*2)         /* 32 KB B stripe (x2 bufs) */

typedef _Float16 half16 __attribute__((ext_vector_type(16)));
typedef float    f32x8  __attribute__((ext_vector_type(8)));
typedef unsigned int u32x4 __attribute__((ext_vector_type(4)));
typedef int      i32x8  __attribute__((ext_vector_type(8)));
typedef int      i32x4  __attribute__((ext_vector_type(4)));

union H16 { half16 h; uint4 q[2]; };

// ---------------------------------------------------------------------------
// TDM descriptor build + issue (2D tile of 2-byte elements, D# per ISA §8).
// For 1D transfers pass tile1 = 0 (tile_dim1 "0 = unused").
// 6-arg builtin: (uint32x4 g0, int32x8 g1, int32x4 g2, int32x4 g3,
//                 int32x8 g4, i32 cpol)
// ---------------------------------------------------------------------------
__device__ __forceinline__ void tdm_load_lds(unsigned ldsAddr, const void* gptr,
                                             unsigned tile0, unsigned tile1,
                                             unsigned tdim0, unsigned tdim1,
                                             unsigned stride0) {
  unsigned long long ga = (unsigned long long)(uintptr_t)gptr;
  u32x4 g0;
  g0[0] = 1u;                                    // count=1, user, no gather
  g0[1] = ldsAddr;                               // lds byte address
  g0[2] = (unsigned)ga;                          // global_addr[31:0]
  g0[3] = (unsigned)((ga >> 32) & 0x1FFFFFFull)  // global_addr[56:32]
          | (2u << 30);                          // type = 2 ("image")
  i32x8 g1;
  g1[0] = 1 << 16;                               // data_size = 1 -> 2 bytes
  g1[1] = (int)((tdim0 & 0xFFFFu) << 16);        // tensor_dim0[15:0]
  g1[2] = (int)((tdim0 >> 16) & 0xFFFFu)         // tensor_dim0[31:16]
        | (int)((tdim1 & 0xFFFFu) << 16);        // tensor_dim1[15:0]
  g1[3] = (int)((tdim1 >> 16) & 0xFFFFu)         // tensor_dim1[31:16]
        | (int)(tile0 << 16);                    // tile_dim0
  g1[4] = (int)(tile1 & 0xFFFFu);                // tile_dim1 (0 => 1D)
  g1[5] = (int)stride0;                          // tensor_dim0_stride[31:0]
  g1[6] = 0;
  g1[7] = 0;
  i32x4 g2 = {0,0,0,0};
  i32x4 g3 = {0,0,0,0};
  i32x8 g4 = {0,0,0,0,0,0,0,0};
  __builtin_amdgcn_tensor_load_to_lds(g0, g1, g2, g3, g4, 0);
}

// ---------------------------------------------------------------------------
// Kernel 1: row L2-normalize, f32 -> f16, plus reciprocal norms.
// ---------------------------------------------------------------------------
__global__ void __launch_bounds__(256)
nrm_kernel(const float* __restrict__ x, _Float16* __restrict__ xh,
           float* __restrict__ rnorm) {
  int wave = threadIdx.x >> 5, lane = threadIdx.x & 31;
  int r = blockIdx.x * 8 + wave;
  const float4* p = (const float4*)(x + (size_t)r * D_ + lane * 16);
  float4 v[4];
  float ss = 0.f;
#pragma unroll
  for (int t = 0; t < 4; ++t) {
    v[t] = p[t];
    ss += v[t].x*v[t].x + v[t].y*v[t].y + v[t].z*v[t].z + v[t].w*v[t].w;
  }
#pragma unroll
  for (int off = 1; off < 32; off <<= 1) ss += __shfl_xor(ss, off, 32);
  float rn = 1.0f / fmaxf(sqrtf(ss), 1e-8f);
  if (lane == 0) rnorm[r] = rn;
  H16 o;
  _Float16* oh = (_Float16*)&o;
#pragma unroll
  for (int t = 0; t < 4; ++t) {
    oh[t*4+0] = (_Float16)(v[t].x * rn);
    oh[t*4+1] = (_Float16)(v[t].y * rn);
    oh[t*4+2] = (_Float16)(v[t].z * rn);
    oh[t*4+3] = (_Float16)(v[t].w * rn);
  }
  uint4* q = (uint4*)(xh + (size_t)r * D_ + lane * 16);
  q[0] = o.q[0];
  q[1] = o.q[1];
}

// ---------------------------------------------------------------------------
// Kernel 2: fused sim-GEMM + streaming masked logsumexp per anchor row.
// 8 waves; wave w owns rows [blk*128 + w*16, +16) and sweeps all columns in
// 32-wide stripes: two independent WMMA accumulator chains per wave.
// B stripes double-buffered in LDS via TDM (TENSORcnt pipelined 1 deep).
// ---------------------------------------------------------------------------
__global__ void __launch_bounds__(256)
sim_lse_kernel(const _Float16* __restrict__ xh, float* __restrict__ lse) {
  extern __shared__ char smem[];
  _Float16* Ah = (_Float16*)smem;
  const int lane = threadIdx.x & 31;
  const int wave = threadIdx.x >> 5;
  const int j  = lane & 15;       // column-within-tile (B/C/D lane role)
  const int hh = lane >> 4;       // lane half
  const int rBase = blockIdx.x * ROWS_WG + wave * 16;
  const unsigned ldsBase = (unsigned)(uintptr_t)smem;  // low 32 bits = LDS offset

  if (threadIdx.x < 32) {
    // A tile: 2D, 512 elems x 128 rows, row stride 512.
    tdm_load_lds(ldsBase, xh + (size_t)blockIdx.x * ROWS_WG * D_,
                 D_, ROWS_WG, D_, N_, D_);
    // B stripe 0: 1D, 32*512 contiguous f16.
    tdm_load_lds(ldsBase + ABYTES, xh, BCOLS * D_, 0, N_ * D_, 1, N_ * D_);
    __builtin_amdgcn_s_wait_tensorcnt(0);
  }
  __syncthreads();

  float m[8], s[8];
#pragma unroll
  for (int e = 0; e < 8; ++e) { m[e] = -1e30f; s[e] = 0.f; }

  const _Float16* Aw = Ah + (wave * 16 + j) * D_;  // A row M = j for this lane

  for (int ct = 0; ct < N_ / BCOLS; ++ct) {
    if (threadIdx.x < 32) {
      if (ct + 1 < N_ / BCOLS) {
        tdm_load_lds(ldsBase + ABYTES + ((unsigned)(ct + 1) & 1u) * BBYTES,
                     xh + (size_t)(ct + 1) * BCOLS * D_,
                     BCOLS * D_, 0, N_ * D_, 1, N_ * D_);
        __builtin_amdgcn_s_wait_tensorcnt(1);  // stripe(ct) done; (ct+1) in flight
      } else {
        __builtin_amdgcn_s_wait_tensorcnt(0);
      }
    }
    __syncthreads();  // stripe(ct) visible to all waves

    const _Float16* Bs = (const _Float16*)(smem + ABYTES + (ct & 1) * BBYTES);
    const _Float16* Bw0 = Bs + j * D_ + 16 * hh;         // cols [ct*32,    +16)
    const _Float16* Bw1 = Bs + (16 + j) * D_ + 16 * hh;  // cols [ct*32+16, +16)
    f32x8 acc0 = {}, acc1 = {};
#pragma unroll
    for (int kk = 0; kk < D_; kk += 32) {
      H16 a, b0, b1;
      // A 16x32 f16 layout: lane half hh -> K = kk+16*hh .. +15 contiguous
      a.q[0] = *(const uint4*)(Aw + kk + 16 * hh);
      a.q[1] = *(const uint4*)(Aw + kk + 16 * hh + 8);
      // B 32x16 f16 layout: lane holds K = kk+16*hh .. +15 contiguous
      b0.q[0] = *(const uint4*)(Bw0 + kk);
      b0.q[1] = *(const uint4*)(Bw0 + kk + 8);
      b1.q[0] = *(const uint4*)(Bw1 + kk);
      b1.q[1] = *(const uint4*)(Bw1 + kk + 8);
      acc0 = __builtin_amdgcn_wmma_f32_16x16x32_f16(false, a.h, false, b0.h,
                                                    (short)0, acc0, false, false);
      acc1 = __builtin_amdgcn_wmma_f32_16x16x32_f16(false, a.h, false, b1.h,
                                                    (short)0, acc1, false, false);
    }

    const int col0 = ct * BCOLS + j;
    const int col1 = col0 + 16;
#pragma unroll
    for (int e = 0; e < 8; ++e) {
      int gRow = rBase + e + 8 * hh;               // C layout: VGPR e, half hh
      float v0 = acc0[e] * INV_TEMP;
      float v1 = acc1[e] * INV_TEMP;
      if (((gRow ^ col0) & (B_ - 1)) == 0) v0 = -__builtin_inff();
      if (((gRow ^ col1) & (B_ - 1)) == 0) v1 = -__builtin_inff();
      float M = fmaxf(m[e], fmaxf(v0, v1));
      s[e] = s[e] * exp2f((m[e] - M) * L2E)
           + exp2f((v0 - M) * L2E) + exp2f((v1 - M) * L2E);
      m[e] = M;
    }
    __syncthreads();  // all waves done reading buffer before it is refilled
  }

  // Merge (m,s) across the 16 lanes of each half (rows 0-7 in lanes 0-15,
  // rows 8-15 in lanes 16-31): xor offsets stay inside the half.
#pragma unroll
  for (int off = 1; off < 16; off <<= 1) {
#pragma unroll
    for (int e = 0; e < 8; ++e) {
      float om = __shfl_xor(m[e], off, 32);
      float os = __shfl_xor(s[e], off, 32);
      float M = fmaxf(m[e], om);
      s[e] = s[e] * exp2f((m[e] - M) * L2E) + os * exp2f((om - M) * L2E);
      m[e] = M;
    }
  }
  if (j == 0) {
#pragma unroll
    for (int e = 0; e < 8; ++e)
      lse[rBase + e + 8 * hh] = m[e] + log2f(s[e]) * LN2;
  }
}

// ---------------------------------------------------------------------------
// Kernel 3: exact f32 positive similarities, one wave per (i,j,b).
// ---------------------------------------------------------------------------
__global__ void __launch_bounds__(256)
pos_kernel(const float* __restrict__ x, const float* __restrict__ rnorm,
           float* __restrict__ pos) {
  int lane = threadIdx.x & 31;
  int idx = blockIdx.x * 8 + (threadIdx.x >> 5);   // 0 .. V*V*B-1
  int i  = idx >> 13;
  int jv = (idx >> 11) & 3;
  int b  = idx & (B_ - 1);
  const float* r0 = x + ((size_t)i  * B_ + b) * D_;
  const float* r1 = x + ((size_t)jv * B_ + b) * D_;
  float dot = 0.f;
#pragma unroll
  for (int t = 0; t < 16; ++t) dot += r0[lane + 32 * t] * r1[lane + 32 * t];
#pragma unroll
  for (int off = 1; off < 32; off <<= 1) dot += __shfl_xor(dot, off, 32);
  if (lane == 0)
    pos[idx] = dot * rnorm[i * B_ + b] * rnorm[jv * B_ + b] * INV_TEMP;
}

// ---------------------------------------------------------------------------
// Kernel 4: deterministic final reduction.
// ---------------------------------------------------------------------------
__global__ void __launch_bounds__(256)
loss_kernel(const float* __restrict__ pos, const float* __restrict__ lse,
            float* __restrict__ out) {
  __shared__ float red[256];
  const int TOT = V_ * (V_ - 1) * B_;  // 24576
  float acc = 0.f;
  for (int t = threadIdx.x; t < TOT; t += 256) {
    int b  = t & (B_ - 1);
    int pi = t >> 11;                 // 0..11 ordered pairs (i, j != i)
    int i  = pi / 3, jj = pi % 3;
    int jv = jj + (jj >= i ? 1 : 0);
    float p = pos[((i * 4 + jv) << 11) + b];
    float l = lse[(i << 11) + b];
    float M = fmaxf(p, l);
    float lae = M + log2f(exp2f((p - M) * L2E) + exp2f((l - M) * L2E)) * LN2;
    acc += lae - p;
  }
  red[threadIdx.x] = acc;
  __syncthreads();
  for (int off = 128; off > 0; off >>= 1) {
    if (threadIdx.x < off) red[threadIdx.x] += red[threadIdx.x + off];
    __syncthreads();
  }
  if (threadIdx.x == 0) out[0] = red[0] / (float)B_;
}

// ---------------------------------------------------------------------------
extern "C" void kernel_launch(void* const* d_in, const int* in_sizes, int n_in,
                              void* d_out, int out_size, void* d_ws, size_t ws_size,
                              hipStream_t stream) {
  const float* x = (const float*)d_in[0];
  char* ws = (char*)d_ws;
  _Float16* xh  = (_Float16*)ws;                       // 8 MB normalized f16
  float* rnorm  = (float*)(ws + (size_t)N_ * D_ * 2);  // 32 KB
  float* lse    = rnorm + N_;                          // 32 KB
  float* pos    = lse + N_;                            // 128 KB
  (void)in_sizes; (void)n_in; (void)out_size; (void)ws_size;

  nrm_kernel<<<N_ / 8, 256, 0, stream>>>(x, xh, rnorm);
  sim_lse_kernel<<<N_ / ROWS_WG, 256, ABYTES + 2 * BBYTES, stream>>>(xh, lse);
  pos_kernel<<<(V_ * V_ * B_) / 8, 256, 0, stream>>>(x, rnorm, pos);
  loss_kernel<<<1, 256, 0, stream>>>(pos, lse, (float*)d_out);
}
